// SparseMoeBlockHFMixtral_32701880992207
// MI455X (gfx1250) — matrix operand order repaired and checked by reading, not compiled, not run on hardware
//
#include <hip/hip_runtime.h>

// ---------------------------------------------------------------------------
// Mixtral-style sparse MoE block for MI455X (gfx1250, wave32, WMMA).
//   B=4, S=2048 -> T=8192 tokens, D=2048, H=4096, E=8, top-2 routing.
// Pipeline: router -> deterministic per-expert compaction -> grouped
// bf16-WMMA GEMMs (gate+up fused w/ SiLU epilogue, then down-proj with
// routing-weight scaling via float atomics; exactly 2 contributions per
// output element -> commutative -> deterministic).
// GEMM2 stages its (already-bf16) A tile via GLOBAL_LOAD_ASYNC_TO_LDS_B128
// (ASYNCcnt path), freeing the VALU pipe for WMMA.
// Workspace use: ~143 MB (combine 256KB, lists 256KB, counters, A buffer
// 17408 x 4096 bf16 = 142.6 MB).
// ---------------------------------------------------------------------------

#define TDIM 8192
#define DDIM 2048
#define HDIM 4096
#define EEXP 8

#define BM 128
#define BN 128
#define BK 64
#define LDK 72   // padded LDS stride (conflict-free for b128 frag loads)

typedef __bf16 bf16;
typedef __attribute__((ext_vector_type(16))) __bf16 v16bf;
typedef __attribute__((ext_vector_type(8)))  __bf16 v8bf;
typedef __attribute__((ext_vector_type(8)))  float   v8f;
typedef __attribute__((ext_vector_type(4)))  float   f32x4;
typedef __attribute__((ext_vector_type(4)))  int     i32x4;

typedef __attribute__((address_space(1))) i32x4* gvec_p;   // global int4*
typedef __attribute__((address_space(3))) i32x4* lvec_p;   // LDS int4*

static __device__ __forceinline__ v16bf ld_frag(const bf16* p0, const bf16* p1) {
  v8bf a = *(const v8bf*)p0;
  v8bf b = *(const v8bf*)p1;
  return __builtin_shufflevector(a, b, 0,1,2,3,4,5,6,7,8,9,10,11,12,13,14,15);
}

static __device__ __forceinline__ v8f zero8() {
  v8f z = {0.f,0.f,0.f,0.f,0.f,0.f,0.f,0.f};
  return z;
}

// 16-byte global -> LDS copy. Prefer the CDNA5 async-to-LDS engine
// (tracked by ASYNCcnt); fall back to a VGPR round-trip if the builtin is
// not declared by this toolchain.
static __device__ __forceinline__ void copy16_g2l(const bf16* g, bf16* l) {
#if __has_builtin(__builtin_amdgcn_global_load_async_to_lds_b128)
  __builtin_amdgcn_global_load_async_to_lds_b128((gvec_p)g, (lvec_p)l, 0, 0);
#else
  *(v8bf*)l = *(const v8bf*)g;
#endif
}

static __device__ __forceinline__ void async_copy_wait() {
#if __has_builtin(__builtin_amdgcn_global_load_async_to_lds_b128)
  asm volatile("s_wait_asynccnt 0x0" ::: "memory");
#endif
}

// ---------------------------------------------------------------- zero out --
__global__ void zero_kernel(f32x4* __restrict__ p, int n) {
  int i = blockIdx.x * blockDim.x + threadIdx.x;
  if (i < n) { f32x4 z = {0.f,0.f,0.f,0.f}; p[i] = z; }
}

// ------------------------------------------------------------------ router --
// One wave per token: lanes partition D, accumulate 8 expert logits,
// shfl-xor reduce, lane 0 does softmax + top-2 + renormalize.
__global__ __launch_bounds__(256) void router_kernel(
    const float* __restrict__ x, const float* __restrict__ wr,
    float* __restrict__ combine)
{
  const int lane = threadIdx.x & 31;
  const int t    = blockIdx.x * 8 + (threadIdx.x >> 5);
  const float* xr = x + (size_t)t * DDIM;

  float acc[EEXP];
#pragma unroll
  for (int e = 0; e < EEXP; ++e) acc[e] = 0.f;

  for (int d = lane; d < DDIM; d += 32) {
    float xv = xr[d];
    const float* w = wr + (size_t)d * EEXP;
#pragma unroll
    for (int e = 0; e < EEXP; ++e) acc[e] = fmaf(xv, w[e], acc[e]);
  }
#pragma unroll
  for (int off = 16; off; off >>= 1)
#pragma unroll
    for (int e = 0; e < EEXP; ++e) acc[e] += __shfl_xor(acc[e], off, 32);

  if (lane == 0) {
    float mx = acc[0];
#pragma unroll
    for (int e = 1; e < EEXP; ++e) mx = fmaxf(mx, acc[e]);
    float ex[EEXP];
#pragma unroll
    for (int e = 0; e < EEXP; ++e) ex[e] = __expf(acc[e] - mx);
    int i0 = 0;
#pragma unroll
    for (int e = 1; e < EEXP; ++e) if (ex[e] > ex[i0]) i0 = e;       // ties -> lowest idx
    int i1 = (i0 == 0) ? 1 : 0;
#pragma unroll
    for (int e = 0; e < EEXP; ++e)
      if (e != i0 && ex[e] > ex[i1]) i1 = e;
    float inv = 1.f / (ex[i0] + ex[i1]);
    float cw[EEXP];
#pragma unroll
    for (int e = 0; e < EEXP; ++e) cw[e] = 0.f;
    cw[i0] = ex[i0] * inv;
    cw[i1] = ex[i1] * inv;
#pragma unroll
    for (int e = 0; e < EEXP; ++e) combine[(size_t)t * EEXP + e] = cw[e];
  }
}

// -------------------------------------------------------------- compaction --
// One workgroup per expert. Deterministic (token-order) stream compaction
// via ballot + block scan. Lists padded to BM with -1 sentinels.
__global__ __launch_bounds__(256) void compact_kernel(
    const float* __restrict__ combine, int* __restrict__ list,
    int* __restrict__ pcount)
{
  const int e    = blockIdx.x;
  const int tid  = threadIdx.x;
  const int lane = tid & 31;
  const int wid  = tid >> 5;

  __shared__ int wsum[8];
  __shared__ int woff[8];
  __shared__ int base;
  __shared__ int ctot;
  __shared__ int cnt_s, pad_s;

  if (tid == 0) base = 0;
  __syncthreads();

  for (int c0 = 0; c0 < TDIM; c0 += 256) {
    int t = c0 + tid;
    int pred = combine[(size_t)t * EEXP + e] > 0.0f ? 1 : 0;
    unsigned long long m = __ballot(pred);
    int rank = __popcll(m & ((1ull << lane) - 1ull));
    int tot  = __popcll(m);
    if (lane == 0) wsum[wid] = tot;
    __syncthreads();
    if (tid == 0) {
      int s = 0;
#pragma unroll
      for (int i = 0; i < 8; ++i) { woff[i] = s; s += wsum[i]; }
      ctot = s;
    }
    __syncthreads();
    if (pred) list[(size_t)e * TDIM + base + woff[wid] + rank] = t;
    __syncthreads();
    if (tid == 0) base += ctot;
    __syncthreads();
  }

  if (tid == 0) {
    cnt_s = base;
    pad_s = ((base + BM - 1) / BM) * BM;
    pcount[e] = pad_s;
  }
  __syncthreads();
  for (int i = cnt_s + tid; i < pad_s; i += 256)
    list[(size_t)e * TDIM + i] = -1;
}

// Exclusive scan of padded counts -> per-expert row base in the A buffer.
__global__ void scan_kernel(const int* __restrict__ pcount, int* __restrict__ rowbase) {
  if (threadIdx.x == 0 && blockIdx.x == 0) {
    int s = 0;
    for (int e = 0; e < EEXP; ++e) { rowbase[e] = s; s += pcount[e]; }
    rowbase[EEXP] = s;
  }
}

// ------------------------------------------------- GEMM1: gate+up + SiLU ---
// A = silu(X @ Wg) * (X @ Wu), X rows gathered per expert. bf16 WMMA,
// fp32 accumulate. 8 waves: 4(m) x 2(n); each wave 32x64 = 2x4 frags.
__global__ __launch_bounds__(256) void gemm1_kernel(
    const float* __restrict__ x,  const float* __restrict__ wg,
    const float* __restrict__ wu, const int* __restrict__ list,
    const int* __restrict__ pcount, const int* __restrict__ rowbase,
    bf16* __restrict__ Abuf)
{
  const int e    = blockIdx.z;
  const int mblk = blockIdx.y;
  const int nblk = blockIdx.x;
  if (mblk * BM >= pcount[e]) return;

  __shared__ bf16 sX[BM][LDK];
  __shared__ bf16 sG[BN][LDK];
  __shared__ bf16 sU[BN][LDK];
  __shared__ int  sTok[BM];

  const int tid  = threadIdx.x;
  const int lane = tid & 31;
  const int wv   = tid >> 5;
  const int wm   = wv >> 1;       // 0..3
  const int wn   = wv & 1;        // 0..1
  const int lrow = lane & 15;
  const int hi   = lane >> 4;     // 0 or 1

  for (int i = tid; i < BM; i += 256) {
    int t = list[(size_t)e * TDIM + mblk * BM + i];
    sTok[i] = t < 0 ? 0 : t;
  }

  v8f accG[2][4], accU[2][4];
#pragma unroll
  for (int i = 0; i < 2; ++i)
#pragma unroll
    for (int j = 0; j < 4; ++j) { accG[i][j] = zero8(); accU[i][j] = zero8(); }

  const float* wgE = wg + (size_t)e * DDIM * HDIM + (size_t)nblk * BN;
  const float* wuE = wu + (size_t)e * DDIM * HDIM + (size_t)nblk * BN;

  for (int k0 = 0; k0 < DDIM; k0 += BK) {
    __syncthreads();
    { // stage X tile (fp32 -> bf16), rows gathered through token list
      int rowi = tid >> 4, c4 = (tid & 15) << 2;
#pragma unroll
      for (int p = 0; p < 8; ++p) {
        int row = p * 16 + rowi;
        f32x4 v = *(const f32x4*)(x + (size_t)sTok[row] * DDIM + k0 + c4);
        bf16* d = &sX[row][c4];
        d[0] = (bf16)v[0]; d[1] = (bf16)v[1]; d[2] = (bf16)v[2]; d[3] = (bf16)v[3];
      }
    }
    { // stage Wg/Wu tiles transposed (fp32 -> bf16), coalesced along N
      int kk = tid >> 5, n4 = (tid & 31) << 2;
#pragma unroll
      for (int p = 0; p < 8; ++p) {
        int k = p * 8 + kk;
        f32x4 g = *(const f32x4*)(wgE + (size_t)(k0 + k) * HDIM + n4);
        f32x4 u = *(const f32x4*)(wuE + (size_t)(k0 + k) * HDIM + n4);
        if (k0 + BK < DDIM) {
          __builtin_prefetch(wgE + (size_t)(k0 + BK + k) * HDIM + n4, 0, 1);
          __builtin_prefetch(wuE + (size_t)(k0 + BK + k) * HDIM + n4, 0, 1);
        }
#pragma unroll
        for (int j = 0; j < 4; ++j) {
          sG[n4 + j][k] = (bf16)g[j];
          sU[n4 + j][k] = (bf16)u[j];
        }
      }
    }
    __syncthreads();

#pragma unroll
    for (int kk = 0; kk < BK; kk += 32) {
      v16bf aF[2];
#pragma unroll
      for (int mf = 0; mf < 2; ++mf) {
        const bf16* r = &sX[wm * 32 + mf * 16 + lrow][0];
        int b = kk + (hi ? 8 : 0);               // A 16-bit layout: K {b..b+7, b+16..b+23}
        aF[mf] = ld_frag(r + b, r + b + 16);
      }
#pragma unroll
      for (int nf = 0; nf < 4; ++nf) {
        const bf16* cg = &sG[wn * 64 + nf * 16 + lrow][kk + (hi ? 8 : 0)];
        const bf16* cu = &sU[wn * 64 + nf * 16 + lrow][kk + (hi ? 8 : 0)];
        v16bf bG = ld_frag(cg, cg + 16);         // B layout: K {b..b+7, b+16..b+23}, same as A
        v16bf bU = ld_frag(cu, cu + 16);
#pragma unroll
        for (int mf = 0; mf < 2; ++mf) {
          accG[mf][nf] = __builtin_amdgcn_wmma_f32_16x16x32_bf16(
              false, aF[mf], false, bG, (short)0, accG[mf][nf], false, false);
          accU[mf][nf] = __builtin_amdgcn_wmma_f32_16x16x32_bf16(
              false, aF[mf], false, bU, (short)0, accU[mf][nf], false, false);
        }
      }
    }
  }

  // epilogue: A = silu(G) * U, store bf16
  const int rbase = rowbase[e] + mblk * BM;
#pragma unroll
  for (int mf = 0; mf < 2; ++mf)
#pragma unroll
    for (int nf = 0; nf < 4; ++nf)
#pragma unroll
      for (int r = 0; r < 8; ++r) {
        int m = wm * 32 + mf * 16 + r + (hi ? 8 : 0);
        int n = nblk * BN + wn * 64 + nf * 16 + lrow;
        float g = accG[mf][nf][r], u = accU[mf][nf][r];
        float a = (g / (1.0f + __expf(-g))) * u;
        Abuf[(size_t)(rbase + m) * HDIM + n] = (bf16)a;
      }
}

// ------------------------------------------------- GEMM2: down projection --
// out[t] += w_route * (A @ Wd). Exactly two real atomic contributions per
// output element (top-2) -> order-independent float result.
// A tile staged via async global->LDS (no conversion needed).
__global__ __launch_bounds__(256) void gemm2_kernel(
    const bf16* __restrict__ Abuf, const float* __restrict__ wd,
    const int* __restrict__ list, const int* __restrict__ pcount,
    const int* __restrict__ rowbase, const float* __restrict__ combine,
    float* __restrict__ out)
{
  const int e    = blockIdx.z;
  const int mblk = blockIdx.y;
  const int nblk = blockIdx.x;
  if (mblk * BM >= pcount[e]) return;

  __shared__ bf16  sA[BM][LDK];
  __shared__ bf16  sW[BN][LDK];
  __shared__ int   sTok[BM];
  __shared__ float sWt[BM];

  const int tid  = threadIdx.x;
  const int lane = tid & 31;
  const int wv   = tid >> 5;
  const int wm   = wv >> 1;
  const int wn   = wv & 1;
  const int lrow = lane & 15;
  const int hi   = lane >> 4;

  for (int i = tid; i < BM; i += 256) {
    int t = list[(size_t)e * TDIM + mblk * BM + i];
    sTok[i] = t < 0 ? 0 : t;
    sWt[i]  = t < 0 ? 0.f : combine[(size_t)t * EEXP + e];
  }

  v8f acc[2][4];
#pragma unroll
  for (int i = 0; i < 2; ++i)
#pragma unroll
    for (int j = 0; j < 4; ++j) acc[i][j] = zero8();

  const size_t abase = (size_t)rowbase[e] + (size_t)mblk * BM;
  const float* wdE = wd + (size_t)e * HDIM * DDIM + (size_t)nblk * BN;

  for (int k0 = 0; k0 < HDIM; k0 += BK) {
    __syncthreads();
    { // stage A tile (already bf16): async global->LDS, 16B per lane
      int rowi = tid >> 3, c8 = (tid & 7) << 3;
#pragma unroll
      for (int p = 0; p < 4; ++p) {
        int row = p * 32 + rowi;
        copy16_g2l(Abuf + (abase + row) * HDIM + k0 + c8, &sA[row][c8]);
      }
    }
    { // stage Wd tile transposed (fp32 -> bf16)
      int kk = tid >> 5, n4 = (tid & 31) << 2;
#pragma unroll
      for (int p = 0; p < 8; ++p) {
        int k = p * 8 + kk;
        f32x4 v = *(const f32x4*)(wdE + (size_t)(k0 + k) * DDIM + n4);
        if (k0 + BK < HDIM)
          __builtin_prefetch(wdE + (size_t)(k0 + BK + k) * DDIM + n4, 0, 1);
#pragma unroll
        for (int j = 0; j < 4; ++j) sW[n4 + j][k] = (bf16)v[j];
      }
    }
    async_copy_wait();    // this wave's async-to-LDS transfers complete
    __syncthreads();      // ... and every wave's tiles are visible

#pragma unroll
    for (int kk = 0; kk < BK; kk += 32) {
      v16bf aF[2];
#pragma unroll
      for (int mf = 0; mf < 2; ++mf) {
        const bf16* r = &sA[wm * 32 + mf * 16 + lrow][0];
        int b = kk + (hi ? 8 : 0);
        aF[mf] = ld_frag(r + b, r + b + 16);
      }
#pragma unroll
      for (int nf = 0; nf < 4; ++nf) {
        const bf16* cw = &sW[wn * 64 + nf * 16 + lrow][kk + (hi ? 8 : 0)];
        v16bf bF = ld_frag(cw, cw + 16);
#pragma unroll
        for (int mf = 0; mf < 2; ++mf)
          acc[mf][nf] = __builtin_amdgcn_wmma_f32_16x16x32_bf16(
              false, aF[mf], false, bF, (short)0, acc[mf][nf], false, false);
      }
    }
  }

  // epilogue: scale by routing weight, accumulate into out with f32 atomics
#pragma unroll
  for (int mf = 0; mf < 2; ++mf)
#pragma unroll
    for (int nf = 0; nf < 4; ++nf)
#pragma unroll
      for (int r = 0; r < 8; ++r) {
        int m = wm * 32 + mf * 16 + r + (hi ? 8 : 0);
        int n = nblk * BN + wn * 64 + nf * 16 + lrow;
        int t = sTok[m];
        float v = sWt[m] * acc[mf][nf][r];
        __hip_atomic_fetch_add(&out[(size_t)t * DDIM + n], v,
                               __ATOMIC_RELAXED, __HIP_MEMORY_SCOPE_AGENT);
      }
}

// ------------------------------------------------------------------ launch --
extern "C" void kernel_launch(void* const* d_in, const int* in_sizes, int n_in,
                              void* d_out, int out_size, void* d_ws, size_t ws_size,
                              hipStream_t stream) {
  const float* x  = (const float*)d_in[0];  // [B,S,D]
  const float* wr = (const float*)d_in[1];  // [D,E]
  const float* wg = (const float*)d_in[2];  // [E,D,H]
  const float* wu = (const float*)d_in[3];  // [E,D,H]
  const float* wd = (const float*)d_in[4];  // [E,H,D]
  float* out = (float*)d_out;               // [B,S,D]

  // Workspace layout (requires ~143 MB):
  char* ws = (char*)d_ws;
  float* combine = (float*)ws;                          // T*E f32   = 256 KB
  int*   list    = (int*)(ws + (size_t)262144);         // E*T i32   = 256 KB
  int*   pcount  = (int*)(ws + (size_t)524288);         // E i32
  int*   rowbase = (int*)(ws + (size_t)524544);         // E+1 i32
  bf16*  Abuf    = (bf16*)(ws + (size_t)528384);        // 17408*H bf16 = 142.6 MB

  zero_kernel<<<(TDIM * DDIM / 4 + 255) / 256, 256, 0, stream>>>(
      (f32x4*)out, TDIM * DDIM / 4);
  router_kernel<<<TDIM / 8, 256, 0, stream>>>(x, wr, combine);
  compact_kernel<<<EEXP, 256, 0, stream>>>(combine, list, pcount);
  scan_kernel<<<1, 32, 0, stream>>>(pcount, rowbase);
  gemm1_kernel<<<dim3(HDIM / BN, TDIM / BM, EEXP), 256, 0, stream>>>(
      x, wg, wu, list, pcount, rowbase, Abuf);
  gemm2_kernel<<<dim3(DDIM / BN, TDIM / BM, EEXP), 256, 0, stream>>>(
      Abuf, wd, list, pcount, rowbase, combine, out);
}
